// SparseGGNNBlock_8126078124646
// MI455X (gfx1250) — matrix operand order repaired and checked by reading, not compiled, not run on hardware
//
#include <hip/hip_runtime.h>
#include <hip/hip_bf16.h>

// ---------------------------------------------------------------------------
// Problem constants (from reference): B=16, N=4096, E=65536, IN=128,
// filters [256,256,128]. Weights shared across batch => each layer GEMM is
// a single [M=65536 x K] @ [K x D] matmul.
// ---------------------------------------------------------------------------
#define B_CONST 16
#define N_CONST 4096
#define E_CONST 65536
#define M_ROWS  (B_CONST * N_CONST)   // 65536

#define M_TILE 128      // rows per block (8 waves x 16)
#define N_TILE 64       // cols per block (4 wmma tiles of 16)
#define K_STEP 32       // bf16 wmma K
#define LDA    40       // bf16 LDS row stride (elements): 80B, 16B-aligned
#define LDAF   36       // f32  LDS row stride (elements): 144B, 16B-aligned

typedef __attribute__((ext_vector_type(16))) __bf16 v16bf;
typedef __attribute__((ext_vector_type(8)))  __bf16 v8bf;
typedef __attribute__((ext_vector_type(8)))  float  v8f;
typedef int gcn_v4i __attribute__((__vector_size__(16)));   // builtin param type

// ---------------------------------------------------------------------------
// CDNA5 async global->LDS copy (ASYNCcnt path), with safe fallback.
// ---------------------------------------------------------------------------
#if defined(__has_builtin)
#  if __has_builtin(__builtin_amdgcn_global_load_async_to_lds_b128)
#    define HAVE_ASYNC_LDS 1
#  endif
#endif
#ifndef HAVE_ASYNC_LDS
#  define HAVE_ASYNC_LDS 0
#endif

__device__ __forceinline__ void async_copy_b128(const float* g, float* l) {
#if HAVE_ASYNC_LDS
    __builtin_amdgcn_global_load_async_to_lds_b128(
        (__attribute__((address_space(1))) gcn_v4i*)g,
        (__attribute__((address_space(3))) gcn_v4i*)l,
        0, 0);
#else
    *(float4*)l = *(const float4*)g;
#endif
}

__device__ __forceinline__ void wait_async_lds() {
#if HAVE_ASYNC_LDS
#  if __has_builtin(__builtin_amdgcn_s_wait_asynccnt)
    __builtin_amdgcn_s_wait_asynccnt(0);
#  else
    asm volatile("s_wait_asynccnt 0" ::: "memory");
#  endif
#endif
}

// ---------------------------------------------------------------------------
// WMMA helpers
// ---------------------------------------------------------------------------
__device__ __forceinline__ v8f wmma_bf16(v16bf a, v16bf b, v8f c) {
    // D(f32 16x16) = A(bf16 16x32) * B(bf16 32x16) + C
    return __builtin_amdgcn_wmma_f32_16x16x32_bf16(
        /*neg_a=*/false, a, /*neg_b=*/false, b,
        /*c_mod=*/(short)0, c, /*reuse_a=*/false, /*reuse_b=*/false);
}

// A-matrix fragment from an fp32 LDS row, converting to bf16 hi/lo in regs.
// Layout (ISA 7.12.2, 16-bit A 16x32): lanes 0-15 (M=lane): K = 0..7,16..23;
// lanes 16-31 (M=lane-16): K = 8..15,24..31.  a = half*8.
__device__ __forceinline__ void frag_a_f32(const float* rowp, int half,
                                           v16bf& fh, v16bf& fl)
{
    const int a = half * 8;
    float xs[16];
    *(float4*)&xs[0]  = *(const float4*)(rowp + a);        // K = a..a+3
    *(float4*)&xs[4]  = *(const float4*)(rowp + a + 4);    // K = a+4..a+7
    *(float4*)&xs[8]  = *(const float4*)(rowp + a + 16);   // K = a+16..a+19
    *(float4*)&xs[12] = *(const float4*)(rowp + a + 20);   // K = a+20..a+23
#pragma unroll
    for (int i = 0; i < 16; ++i) {
        __bf16 h = (__bf16)xs[i];
        fh[i] = h;
        fl[i] = (__bf16)(xs[i] - (float)h);
    }
}

// B-matrix fragment (32x16 bf16), rowp points at Wt[n][0] (K contiguous):
// lanes 0-15 (N=lane) hold K = 0..7,16..23; lanes 16-31 hold K = 8..15,24..31.
__device__ __forceinline__ v16bf frag_b(const __bf16* rowp, int half) {
    const int kb = half * 8;
    v8bf c0 = *(const v8bf*)(rowp + kb);
    v8bf c1 = *(const v8bf*)(rowp + kb + 16);
    v16bf f;
#pragma unroll
    for (int i = 0; i < 8; ++i) { f[i] = c0[i]; f[i + 8] = c1[i]; }
    return f;
}

// ---------------------------------------------------------------------------
// Tile staging helpers
// ---------------------------------------------------------------------------
// A tile: 128x32 f32 (16 KB) copied raw into LDS; 4 x b128 per thread.
__device__ __forceinline__ void load_a_tile(float* dst, const float* __restrict__ X,
                                            int m0, int K, int k0, int tid)
{
#pragma unroll
    for (int i = 0; i < 4; ++i) {
        int linear = tid + 256 * i;        // float4 index, 1024 total
        int r  = linear >> 3;              // 8 float4 per row
        int c4 = linear & 7;
        async_copy_b128(X + (size_t)(m0 + r) * K + k0 + c4 * 4,
                        dst + r * LDAF + c4 * 4);
    }
}

// W tile: 32x64 f32 -> transposed bf16 hi/lo LDS Wt[n][k].
// Thread owns (n = tid&63, khalf = tid>>6): 8 consecutive k for one n.
// Global reads are lane-coalesced over n; LDS writes are 2 x ds_store_b128.
__device__ __forceinline__ void stage_b_tile(__bf16* bhi, __bf16* blo,
                                             const float* __restrict__ W,
                                             int n0, int Nout, int k0, int tid)
{
    const int n  = tid & 63;
    const int kh = tid >> 6;               // 0..3
    float xs[8];
#pragma unroll
    for (int j = 0; j < 8; ++j)
        xs[j] = W[(size_t)(k0 + kh * 8 + j) * Nout + n0 + n];
    v8bf hv, lv;
#pragma unroll
    for (int j = 0; j < 8; ++j) {
        __bf16 h = (__bf16)xs[j];
        hv[j] = h;
        lv[j] = (__bf16)(xs[j] - (float)h);
    }
    *(v8bf*)(bhi + n * LDA + kh * 8) = hv;
    *(v8bf*)(blo + n * LDA + kh * 8) = lv;
}

// ---------------------------------------------------------------------------
// Split-bf16 GEMM:  H = X @ W  (x ~ x_hi + x_lo; 3 WMMAs per tile pair for
// ~fp32 accuracy with fp32 accumulation).  Double-buffered: the next A tile
// streams global->LDS asynchronously while the current tiles feed the WMMAs.
// Rows with mask[row]==0 are zeroed in the epilogue.
// grid = (Nout/N_TILE, M_ROWS/M_TILE), block = 256 threads (8 wave32).
// ---------------------------------------------------------------------------
__global__ __launch_bounds__(256)
void gcn_gemm_bf16x2(const float* __restrict__ X,
                     const float* __restrict__ W,
                     const int*   __restrict__ mask,
                     float*       __restrict__ H,
                     int K, int Nout)
{
    __shared__ __align__(16) float  Asf[2][M_TILE * LDAF];   // 36.0 KB
    __shared__ __align__(16) __bf16 Bs_hi[2][N_TILE * LDA];  // 10.0 KB
    __shared__ __align__(16) __bf16 Bs_lo[2][N_TILE * LDA];  // 10.0 KB

    const int tid  = threadIdx.x;
    const int wave = tid >> 5;
    const int lane = tid & 31;
    const int half = lane >> 4;
    const int nlan = lane & 15;

    const int m0 = blockIdx.y * M_TILE;
    const int n0 = blockIdx.x * N_TILE;

    v8f acc[4];
#pragma unroll
    for (int nt = 0; nt < 4; ++nt)
        acc[nt] = (v8f){0.f, 0.f, 0.f, 0.f, 0.f, 0.f, 0.f, 0.f};

    // Prologue: stage tile 0.
    load_a_tile(Asf[0], X, m0, K, 0, tid);
    stage_b_tile(Bs_hi[0], Bs_lo[0], W, n0, Nout, 0, tid);
    wait_async_lds();
    __syncthreads();

    const int nsteps = K >> 5;
    for (int s = 0; s < nsteps; ++s) {
        const int cbuf = s & 1;
        const int nbuf = cbuf ^ 1;

        // Kick off next A tile copy (overlaps with WMMA below).
        if (s + 1 < nsteps)
            load_a_tile(Asf[nbuf], X, m0, K, (s + 1) * K_STEP, tid);

        // Compute on current buffers.
        const float* arow = Asf[cbuf] + (wave * 16 + nlan) * LDAF;
        v16bf a_hi, a_lo;
        frag_a_f32(arow, half, a_hi, a_lo);
#pragma unroll
        for (int nt = 0; nt < 4; ++nt) {
            v16bf b_hi = frag_b(Bs_hi[cbuf] + (nt * 16 + nlan) * LDA, half);
            v16bf b_lo = frag_b(Bs_lo[cbuf] + (nt * 16 + nlan) * LDA, half);
            acc[nt] = wmma_bf16(a_hi, b_hi, acc[nt]);
            acc[nt] = wmma_bf16(a_hi, b_lo, acc[nt]);
            acc[nt] = wmma_bf16(a_lo, b_hi, acc[nt]);
        }

        // Stage next B tile (writes the other buffer; no hazard pre-barrier).
        if (s + 1 < nsteps)
            stage_b_tile(Bs_hi[nbuf], Bs_lo[nbuf], W, n0, Nout,
                         (s + 1) * K_STEP, tid);

        wait_async_lds();
        __syncthreads();
    }

    // Epilogue: D element (vgpr r, lane l) -> (M = r + 8*half, N = l&15).
    const int mbase = m0 + wave * 16 + 8 * half;
    float mskv[8];
#pragma unroll
    for (int r = 0; r < 8; ++r)
        mskv[r] = (mask[mbase + r] > 0) ? 1.f : 0.f;
#pragma unroll
    for (int nt = 0; nt < 4; ++nt) {
        int col = n0 + nt * 16 + nlan;
#pragma unroll
        for (int r = 0; r < 8; ++r)
            H[(size_t)(mbase + r) * Nout + col] = acc[nt][r] * mskv[r];
    }
}

// ---------------------------------------------------------------------------
// Zero workspace region (float4 granularity).
// ---------------------------------------------------------------------------
__global__ __launch_bounds__(256)
void gcn_zero(float4* __restrict__ p, long n4)
{
    long idx = (long)blockIdx.x * 256 + threadIdx.x;
    if (idx < n4) p[idx] = make_float4(0.f, 0.f, 0.f, 0.f);
}

// ---------------------------------------------------------------------------
// Edge scatter-add: one wave per edge; lanes stride channels.
// agg[b][v][:] += ew * h[b][u][:]   (invalid-u rows of h are already zero;
// invalid-v rows are zeroed by the mask epilogue).  h and agg both fit in
// the 192 MB L2, so the random gathers/atomics run at L2 speed.
// ---------------------------------------------------------------------------
__global__ __launch_bounds__(256)
void gcn_scatter(const float* __restrict__ H,
                 const int*   __restrict__ ei,
                 const float* __restrict__ ew,
                 float*       __restrict__ AGG,
                 int D)
{
    const int g    = blockIdx.x * 8 + (threadIdx.x >> 5);  // edge id in [0, B*E)
    const int lane = threadIdx.x & 31;
    const int b    = g >> 16;          // E = 65536
    const int e    = g & (E_CONST - 1);

    const int* eib = ei + (size_t)b * 2 * E_CONST;         // wave-uniform scalars
    const int   u  = eib[e];
    const int   v  = eib[E_CONST + e];
    const float w  = ew[(size_t)b * E_CONST + e];

    const float* hs = H   + ((size_t)b * N_CONST + u) * D;
    float*       as = AGG + ((size_t)b * N_CONST + v) * D;
#pragma unroll 4
    for (int c = lane; c < D; c += 32) {
        (void)__hip_atomic_fetch_add(&as[c], w * hs[c],
                                     __ATOMIC_RELAXED, __HIP_MEMORY_SCOPE_AGENT);
    }
}

// ---------------------------------------------------------------------------
// out = valid(row) ? relu(agg + bias) : 0
// ---------------------------------------------------------------------------
__global__ __launch_bounds__(256)
void gcn_bias_relu_mask(const float* __restrict__ AGG,
                        const float* __restrict__ bias,
                        const int*   __restrict__ mask,
                        float*       __restrict__ OUT,
                        int D, int log2d)
{
    size_t idx = (size_t)blockIdx.x * 256 + threadIdx.x;   // over M_ROWS*D
    int row = (int)(idx >> log2d);
    int c   = (int)(idx & (size_t)(D - 1));
    float v = AGG[idx] + bias[c];
    v = v > 0.f ? v : 0.f;
    OUT[idx] = (mask[row] > 0) ? v : 0.f;
}

// ---------------------------------------------------------------------------
// Host-side orchestration. Workspace layout (floats):
//   P0  [M_ROWS*256]   GEMM output h
//   P1  [M_ROWS*256]   post-activation x for next layer
//   AGG [M_ROWS*256]   scatter accumulator
// Total 3 * 16 Mi floats = ~201 MB of d_ws.
// ---------------------------------------------------------------------------
extern "C" void kernel_launch(void* const* d_in, const int* in_sizes, int n_in,
                              void* d_out, int out_size, void* d_ws, size_t ws_size,
                              hipStream_t stream)
{
    const float* x    = (const float*)d_in[0];
    const int*   ei   = (const int*)  d_in[1];
    const float* ew   = (const float*)d_in[2];
    const int*   mask = (const int*)  d_in[3];
    const float* Wl[3] = {(const float*)d_in[4], (const float*)d_in[6], (const float*)d_in[8]};
    const float* bl[3] = {(const float*)d_in[5], (const float*)d_in[7], (const float*)d_in[9]};
    float* out = (float*)d_out;

    const size_t SZ = (size_t)M_ROWS * 256;
    float* P0  = (float*)d_ws;
    float* P1  = P0 + SZ;
    float* AGG = P1 + SZ;

    const int Kdims[3] = {128, 256, 256};
    const int Ddims[3] = {256, 256, 128};
    const float* xin = x;

    for (int layer = 0; layer < 3; ++layer) {
        const int K = Kdims[layer];
        const int D = Ddims[layer];
        float* dst = (layer == 2) ? out : P1;

        dim3 gg(D / N_TILE, M_ROWS / M_TILE);
        gcn_gemm_bf16x2<<<gg, dim3(256), 0, stream>>>(xin, Wl[layer], mask, P0, K, D);

        long n4 = (long)M_ROWS * D / 4;
        gcn_zero<<<(unsigned)((n4 + 255) / 256), 256, 0, stream>>>((float4*)AGG, n4);

        gcn_scatter<<<(B_CONST * E_CONST) / 8, 256, 0, stream>>>(P0, ei, ew, AGG, D);

        int log2d = (D == 256) ? 8 : 7;
        size_t ne = (size_t)M_ROWS * D;
        gcn_bias_relu_mask<<<(unsigned)(ne / 256), 256, 0, stream>>>(AGG, bl[layer], mask, dst, D, log2d);

        xin = P1;
    }
}
